// DeepSeekV3_26877905338907
// MI455X (gfx1250) — compile-verified, matrix operand order checked by reading, not run on hardware
//
#include <hip/hip_runtime.h>
#include <hip/hip_bf16.h>
#include <math.h>

#define USE_ASYNC 1   // 1: global_load_async_to_lds_b128 pipeline; 0: sync staging

// ---------------- model dims ----------------
static constexpr int B_   = 2;
static constexpr int S_   = 1024;
static constexpr int T_   = B_ * S_;     // 2048 tokens
static constexpr int H_   = 1024;
static constexpr int NH_  = 16;
static constexpr int KVC_ = 256;
static constexpr int QC_  = 384;
static constexpr int E_   = 8;
static constexpr int I_   = 4096;
static constexpr int L_   = 2;
static constexpr int V_   = 32000;

// ---------------- WMMA types ----------------
typedef __attribute__((ext_vector_type(16))) __bf16          v16bf;
typedef __attribute__((ext_vector_type(8)))  float           v8f;
typedef __attribute__((ext_vector_type(8)))  unsigned short  ushort8v;
typedef __attribute__((ext_vector_type(4)))  unsigned short  ushort4v;

union FragAB { v16bf v; ushort8v h[2]; unsigned short u[16]; };

__device__ __forceinline__ unsigned short f2bf(float f) {
  union { float f; unsigned u; } c; c.f = f;
  unsigned u = c.u;
  return (unsigned short)((u + 0x7FFFu + ((u >> 16) & 1u)) >> 16); // RNE
}

__device__ __forceinline__ float gelu_exact(float x) {
  return 0.5f * x * (1.0f + erff(x * 0.70710678118654752440f));
}

// =====================================================================
// bf16-resident WMMA GEMM:  out (+)= epi(A[M,K](bf16) @ Bt[N,K](bf16)^T)
// Block tile 128x128, K-step 32, 8 waves (4x2), each 32x64 (8 WMMAs).
// LDS tiles [m][k]/[n][k], row stride 48 halfwords (96B, 16B-aligned).
// Double-buffered; tiles DMA'd with GLOBAL_LOAD_ASYNC_TO_LDS_B128.
// flags: 1=gelu, 2=accumulate into f32 C, 4=store f32 C, 8=store bf16 Cb
// =====================================================================
static constexpr int LDT = 48;            // halfwords per LDS row (32 + 16 pad)

#if USE_ASYNC
__device__ __forceinline__ void async_ld_b128(unsigned lds_off, unsigned byte_off,
                                              const void* base) {
  asm volatile("global_load_async_to_lds_b128 %0, %1, %2"
               :: "v"(lds_off), "v"(byte_off), "s"(base) : "memory");
}
__device__ __forceinline__ void wait_async_le4() {
  asm volatile("s_wait_asynccnt 0x4" ::: "memory");
}
__device__ __forceinline__ void wait_async_le0() {
  asm volatile("s_wait_asynccnt 0x0" ::: "memory");
}
#endif

__global__ __launch_bounds__(256) void gemm_bf16_wmma(
    const unsigned short* __restrict__ A,   // [M][K] bf16
    const unsigned short* __restrict__ Bt,  // [N][K] bf16 (pre-transposed)
    const float* __restrict__ bias,
    const float* __restrict__ rowscale, int rs_stride,
    float* __restrict__ C, unsigned short* __restrict__ Cb,
    int M, int N, int K, float postScale, int flags)
{
  __shared__ __align__(16) unsigned short sA[2][128 * LDT];
  __shared__ __align__(16) unsigned short sB[2][128 * LDT];

  const int tid  = threadIdx.x;
  const int lane = tid & 31;
  const int wid  = tid >> 5;
  const int wm   = wid >> 1;     // 0..3  -> 32-row band
  const int wn   = wid & 1;      // 0..1  -> 64-col band
  const int half = lane >> 4;
  const int r    = lane & 15;

  const int m0 = blockIdx.y * 128;
  const int n0 = blockIdx.x * 128;

  v8f acc[2][4];
#pragma unroll
  for (int fi = 0; fi < 2; ++fi)
#pragma unroll
    for (int fj = 0; fj < 4; ++fj) acc[fi][fj] = v8f{};

  // per-thread chunk mapping: 512 chunks of 16B per 128x32 tile, 2 per thread
  const int c0row = (tid + 0)   >> 2, c0k8 = ((tid + 0)   & 3) << 3;
  const int c1row = (tid + 256) >> 2, c1k8 = ((tid + 256) & 3) << 3;

#if USE_ASYNC
  auto issue = [&](int buf, int kk) {
    unsigned lA0 = (unsigned)(size_t)(const void*)(&sA[buf][c0row * LDT + c0k8]);
    unsigned lA1 = (unsigned)(size_t)(const void*)(&sA[buf][c1row * LDT + c1k8]);
    unsigned lB0 = (unsigned)(size_t)(const void*)(&sB[buf][c0row * LDT + c0k8]);
    unsigned lB1 = (unsigned)(size_t)(const void*)(&sB[buf][c1row * LDT + c1k8]);
    async_ld_b128(lA0, (unsigned)(((size_t)(m0 + c0row) * K + kk + c0k8) * 2), A);
    async_ld_b128(lA1, (unsigned)(((size_t)(m0 + c1row) * K + kk + c1k8) * 2), A);
    async_ld_b128(lB0, (unsigned)(((size_t)(n0 + c0row) * K + kk + c0k8) * 2), Bt);
    async_ld_b128(lB1, (unsigned)(((size_t)(n0 + c1row) * K + kk + c1k8) * 2), Bt);
  };
  issue(0, 0);
#else
  auto stage = [&](int buf, int kk) {
    *(ushort8v*)(&sA[buf][c0row * LDT + c0k8]) =
        *(const ushort8v*)(A + (size_t)(m0 + c0row) * K + kk + c0k8);
    *(ushort8v*)(&sA[buf][c1row * LDT + c1k8]) =
        *(const ushort8v*)(A + (size_t)(m0 + c1row) * K + kk + c1k8);
    *(ushort8v*)(&sB[buf][c0row * LDT + c0k8]) =
        *(const ushort8v*)(Bt + (size_t)(n0 + c0row) * K + kk + c0k8);
    *(ushort8v*)(&sB[buf][c1row * LDT + c1k8]) =
        *(const ushort8v*)(Bt + (size_t)(n0 + c1row) * K + kk + c1k8);
  };
  stage(0, 0);
#endif

  int cur = 0;
  for (int kk = 0; kk < K; kk += 32) {
    int nxt = kk + 32;
#if USE_ASYNC
    if (nxt < K) { issue(cur ^ 1, nxt); wait_async_le4(); }
    else         { wait_async_le0(); }
#else
    if (nxt < K) stage(cur ^ 1, nxt);
#endif
    __syncthreads();

    FragAB af[2], bfr[4];
#pragma unroll
    for (int fi = 0; fi < 2; ++fi) {
      const unsigned short* p = &sA[cur][(wm * 32 + fi * 16 + r) * LDT + half * 8];
      af[fi].h[0] = *(const ushort8v*)(p);
      af[fi].h[1] = *(const ushort8v*)(p + 16);
    }
#pragma unroll
    for (int fj = 0; fj < 4; ++fj) {
      const unsigned short* p = &sB[cur][(wn * 64 + fj * 16 + r) * LDT + half * 8];
      bfr[fj].h[0] = *(const ushort8v*)(p);
      bfr[fj].h[1] = *(const ushort8v*)(p + 16);
    }

#pragma unroll
    for (int fi = 0; fi < 2; ++fi)
#pragma unroll
      for (int fj = 0; fj < 4; ++fj)
        acc[fi][fj] = __builtin_amdgcn_wmma_f32_16x16x32_bf16(
            false, af[fi].v, false, bfr[fj].v, (short)0, acc[fi][fj],
            false, false);
    __syncthreads();
    cur ^= 1;
  }

  // ---- epilogue ----
#pragma unroll
  for (int fi = 0; fi < 2; ++fi)
#pragma unroll
    for (int fj = 0; fj < 4; ++fj)
#pragma unroll
      for (int i = 0; i < 8; ++i) {
        int row = m0 + wm * 32 + fi * 16 + half * 8 + i;
        int col = n0 + wn * 64 + fj * 16 + r;
        float v = acc[fi][fj][i] + bias[col];
        if (flags & 1) v = gelu_exact(v);
        v *= postScale;
        if (rowscale) v *= rowscale[(size_t)row * rs_stride];
        size_t off = (size_t)row * N + col;
        if (flags & 2)      C[off] += v;
        else if (flags & 4) C[off] = v;
        if (flags & 8)      Cb[off] = f2bf(v);
      }
}

// =====================================================================
// weight transpose + convert: fp32 W[K][N] -> bf16 Wt[N][K], 64x64 tiles
// =====================================================================
static constexpr int TLD = 72;  // padded halfword stride of transpose tile

__global__ __launch_bounds__(256) void convert_wt_kernel(
    const float* __restrict__ W, unsigned short* __restrict__ Wt, int K, int N) {
  __shared__ unsigned short tile[64 * TLD];
  const int tid = threadIdx.x;
  const int k0  = blockIdx.y * 64;
  const int n0  = blockIdx.x * 64;
#pragma unroll
  for (int i = 0; i < 4; ++i) {
    int q  = tid + i * 256;          // 0..1023 quads
    int k  = q >> 4;                 // 16 quads per 64-wide row
    int n4 = (q & 15) << 2;
    float4 f = *(const float4*)(W + (size_t)(k0 + k) * N + n0 + n4);
    tile[(n4 + 0) * TLD + k] = f2bf(f.x);
    tile[(n4 + 1) * TLD + k] = f2bf(f.y);
    tile[(n4 + 2) * TLD + k] = f2bf(f.z);
    tile[(n4 + 3) * TLD + k] = f2bf(f.w);
  }
  __syncthreads();
#pragma unroll
  for (int i = 0; i < 4; ++i) {
    int q  = tid + i * 256;
    int n  = q >> 4;
    int k4 = (q & 15) << 2;
    ushort4v hv = *(const ushort4v*)(&tile[n * TLD + k4]);
    *(ushort4v*)(Wt + (size_t)(n0 + n) * K + k0 + k4) = hv;
  }
}

// =====================================================================
// elementwise / reduction kernels
// =====================================================================
__global__ void embed_kernel(const int* __restrict__ ids,
                             const float* __restrict__ emb,
                             float* __restrict__ x) {
  int idx = blockIdx.x * 256 + threadIdx.x;
  if (idx >= T_ * H_) return;
  int t = idx >> 10, h = idx & (H_ - 1);
  x[idx] = emb[(size_t)ids[t] * H_ + h];
}

__global__ void fill_zero_kernel(float* __restrict__ p, int n) {
  int idx = blockIdx.x * 256 + threadIdx.x;
  if (idx < n) p[idx] = 0.0f;
}

// LayerNorm -> f32 out + bf16 out
__global__ __launch_bounds__(256) void layernorm_kernel(
    const float* __restrict__ x, const float* __restrict__ w,
    const float* __restrict__ b, float* __restrict__ out,
    unsigned short* __restrict__ outb) {
  __shared__ float rs[256], rq[256];
  int t = blockIdx.x, tid = threadIdx.x;
  const float* xr = x + (size_t)t * H_;
  float s = 0.f, q = 0.f;
  for (int h = tid; h < H_; h += 256) { float v = xr[h]; s += v; q += v * v; }
  rs[tid] = s; rq[tid] = q; __syncthreads();
  for (int off = 128; off > 0; off >>= 1) {
    if (tid < off) { rs[tid] += rs[tid + off]; rq[tid] += rq[tid + off]; }
    __syncthreads();
  }
  float mean = rs[0] * (1.0f / H_);
  float var  = rq[0] * (1.0f / H_) - mean * mean;
  float inv  = rsqrtf(var + 1e-5f);
  for (int h = tid; h < H_; h += 256) {
    float v = (xr[h] - mean) * inv * w[h] + b[h];
    out[(size_t)t * H_ + h]  = v;
    outb[(size_t)t * H_ + h] = f2bf(v);
  }
}

// RoPE: rotates along the HEADS axis (pos = head index), per reference.
__global__ void rope_kernel(float* __restrict__ x) {
  int idx = blockIdx.x * 256 + threadIdx.x;       // t*512 + h*32 + i
  if (idx >= T_ * NH_ * 32) return;
  int i = idx & 31, h = (idx >> 5) & 15, t = idx >> 9;
  float invf = __expf(-((float)(2 * i) / 64.0f) * 9.210340371976184f); // ln 1e4
  float ang = (float)h * invf;
  float sn, cs; __sincosf(ang, &sn, &cs);
  size_t base = (size_t)t * 1024 + h * 64 + 2 * i;
  float x1 = x[base], x2 = x[base + 1];
  x[base]     = x1 * cs - x2 * sn;
  x[base + 1] = x1 * sn + x2 * cs;
}

// softmax attention: one block per (b, h, q); writes bf16 context
__global__ __launch_bounds__(256) void mla_attn_kernel(
    const float* __restrict__ qc, const float* __restrict__ qr,
    const float* __restrict__ kc, const float* __restrict__ kr,
    const float* __restrict__ vv, unsigned short* __restrict__ ctxb) {
  __shared__ float qv[128];
  __shared__ float sc[S_];
  __shared__ float red[256];
  int tid = threadIdx.x;
  int s = blockIdx.x & (S_ - 1);
  int h = (blockIdx.x >> 10) & (NH_ - 1);
  int b = blockIdx.x >> 14;
  size_t tq = (size_t)b * S_ + s;
  if (tid < 64)       qv[tid] = qc[tq * 1024 + h * 64 + tid];
  else if (tid < 128) qv[tid] = qr[tq * 1024 + h * 64 + (tid - 64)];
  __syncthreads();

  float mx = -1e30f;
  for (int k = tid; k < S_; k += 256) {
    size_t tk = (size_t)b * S_ + k;
    const float* kcp = kc + tk * 1024 + h * 64;
    const float* krp = kr + tk * 1024 + h * 64;
    float d = 0.f;
#pragma unroll 8
    for (int j = 0; j < 64; ++j) d += qv[j] * kcp[j];
#pragma unroll 8
    for (int j = 0; j < 64; ++j) d += qv[64 + j] * krp[j];
    d *= 0.08838834764831844f;               // 1/sqrt(HD+RD)
    sc[k] = d;
    mx = fmaxf(mx, d);
  }
  red[tid] = mx; __syncthreads();
  for (int off = 128; off > 0; off >>= 1) {
    if (tid < off) red[tid] = fmaxf(red[tid], red[tid + off]);
    __syncthreads();
  }
  mx = red[0]; __syncthreads();

  float sum = 0.f;
  for (int k = tid; k < S_; k += 256) { float e = __expf(sc[k] - mx); sc[k] = e; sum += e; }
  red[tid] = sum; __syncthreads();
  for (int off = 128; off > 0; off >>= 1) {
    if (tid < off) red[tid] += red[tid + off];
    __syncthreads();
  }
  float inv = 1.0f / red[0]; __syncthreads();

  int d = tid & 63, chunk = tid >> 6;
  float a = 0.f;
  for (int k = chunk * 256; k < chunk * 256 + 256; ++k) {
    size_t tk = (size_t)b * S_ + k;
    a += sc[k] * vv[tk * 1024 + h * 64 + d];
  }
  red[tid] = a; __syncthreads();
  if (tid < 64)
    ctxb[tq * 1024 + h * 64 + tid] =
        f2bf((red[tid] + red[64 + tid] + red[128 + tid] + red[192 + tid]) * inv);
}

// sigmoid gating + top-2 normalize; writes dense gate[T,E] (0 off-topk)
__global__ __launch_bounds__(256) void moe_gate_kernel(
    const float* __restrict__ xn, const float* __restrict__ gw,
    const float* __restrict__ gbias, float* __restrict__ gate) {
  __shared__ float red[E_ * 256];
  int t = blockIdx.x, tid = threadIdx.x;
  float p[E_];
#pragma unroll
  for (int e = 0; e < E_; ++e) p[e] = 0.f;
  for (int h = tid; h < H_; h += 256) {
    float xv = xn[(size_t)t * H_ + h];
#pragma unroll
    for (int e = 0; e < E_; ++e) p[e] += xv * gw[h * E_ + e];
  }
#pragma unroll
  for (int e = 0; e < E_; ++e) red[e * 256 + tid] = p[e];
  __syncthreads();
  for (int off = 128; off > 0; off >>= 1) {
    if (tid < off)
#pragma unroll
      for (int e = 0; e < E_; ++e) red[e * 256 + tid] += red[e * 256 + tid + off];
    __syncthreads();
  }
  if (tid == 0) {
    float s[E_];
#pragma unroll
    for (int e = 0; e < E_; ++e)
      s[e] = 1.0f / (1.0f + __expf(-(red[e * 256] + gbias[e])));
    int i1 = 0;
    for (int e = 1; e < E_; ++e) if (s[e] > s[i1]) i1 = e;
    int i2 = (i1 == 0) ? 1 : 0;
    for (int e = 0; e < E_; ++e) if (e != i1 && s[e] > s[i2]) i2 = e;
    float denom = s[i1] + s[i2] + 1e-6f;
    for (int e = 0; e < E_; ++e)
      gate[(size_t)t * E_ + e] = (e == i1 || e == i2) ? s[e] / denom : 0.f;
  }
}

// out = comb / (maxabs+1e-6); x += out  (residual)
__global__ __launch_bounds__(256) void moe_final_kernel(
    const float* __restrict__ comb, float* __restrict__ x) {
  __shared__ float red[256];
  int t = blockIdx.x, tid = threadIdx.x;
  const float* cr = comb + (size_t)t * H_;
  float m = 0.f;
  for (int h = tid; h < H_; h += 256) m = fmaxf(m, fabsf(cr[h]));
  red[tid] = m; __syncthreads();
  for (int off = 128; off > 0; off >>= 1) {
    if (tid < off) red[tid] = fmaxf(red[tid], red[tid + off]);
    __syncthreads();
  }
  float inv = 1.0f / (red[0] + 1e-6f);
  for (int h = tid; h < H_; h += 256) x[(size_t)t * H_ + h] += cr[h] * inv;
}

// =====================================================================
// host orchestration
// =====================================================================
extern "C" void kernel_launch(void* const* d_in, const int* in_sizes, int n_in,
                              void* d_out, int out_size, void* d_ws, size_t ws_size,
                              hipStream_t stream) {
  (void)in_sizes; (void)n_in; (void)out_size; (void)ws_size;

  const int*   ids     = (const int*)  d_in[0];
  const float* emb     = (const float*)d_in[1];
  const float* anw     = (const float*)d_in[2];
  const float* anb     = (const float*)d_in[3];
  const float* kvdw    = (const float*)d_in[4];
  const float* kvdb    = (const float*)d_in[5];
  const float* kupw    = (const float*)d_in[6];
  const float* kupb    = (const float*)d_in[7];
  const float* vupw    = (const float*)d_in[8];
  const float* vupb    = (const float*)d_in[9];
  const float* krw     = (const float*)d_in[10];
  const float* krb     = (const float*)d_in[11];
  const float* qdw     = (const float*)d_in[12];
  const float* qdb     = (const float*)d_in[13];
  const float* qupw    = (const float*)d_in[14];
  const float* qupb    = (const float*)d_in[15];
  const float* qrw     = (const float*)d_in[16];
  const float* qrb_    = (const float*)d_in[17];
  const float* outw    = (const float*)d_in[18];
  const float* outb    = (const float*)d_in[19];
  const float* mnw     = (const float*)d_in[20];
  const float* mnb     = (const float*)d_in[21];
  const float* gatew   = (const float*)d_in[22];
  const float* moebias = (const float*)d_in[23];
  const float* eupw    = (const float*)d_in[24];
  const float* eupb    = (const float*)d_in[25];
  const float* ednw    = (const float*)d_in[26];
  const float* ednb    = (const float*)d_in[27];
  const float* supw    = (const float*)d_in[28];
  const float* supb    = (const float*)d_in[29];
  const float* sdnw    = (const float*)d_in[30];
  const float* sdnb    = (const float*)d_in[31];
  const float* fnw     = (const float*)d_in[32];
  const float* fnb     = (const float*)d_in[33];
  const float* headw   = (const float*)d_in[34];
  const float* headb   = (const float*)d_in[35];
  float*       logits  = (float*)d_out;

  // bump allocator over workspace (fp32 units; bf16 buffers use half slots)
  float* ws = (float*)d_ws;
  float* x    = ws; ws += (size_t)T_ * H_;
  float* xn   = ws; ws += (size_t)T_ * H_;
  float* kc   = ws; ws += (size_t)T_ * 1024;
  float* vv   = ws; ws += (size_t)T_ * 1024;
  float* kr   = ws; ws += (size_t)T_ * 1024;
  float* qc   = ws; ws += (size_t)T_ * 1024;
  float* qr   = ws; ws += (size_t)T_ * 1024;
  float* comb = ws; ws += (size_t)T_ * H_;
  float* gate = ws; ws += (size_t)T_ * E_;
  unsigned short* xnb   = (unsigned short*)ws; ws += (size_t)T_ * H_ / 2;
  unsigned short* kvb   = (unsigned short*)ws; ws += (size_t)T_ * KVC_ / 2;
  unsigned short* qdb_b = (unsigned short*)ws; ws += (size_t)T_ * QC_ / 2;
  unsigned short* ctxb  = (unsigned short*)ws; ws += (size_t)T_ * 1024 / 2;
  unsigned short* hmidb = (unsigned short*)ws; ws += (size_t)T_ * I_ / 2;
  unsigned short* bfB   = (unsigned short*)ws; ws += (size_t)V_ * H_ / 2;

  auto cvtW = [&](const float* W, int K, int N) {
    dim3 grid(N / 64, K / 64);
    convert_wt_kernel<<<grid, 256, 0, stream>>>(W, bfB, K, N);
  };
  auto gemm = [&](const unsigned short* A, const float* bias,
                  float* C, unsigned short* Cb,
                  int M, int N, int K, float scale, int flags,
                  const float* rowscale, int rs_stride) {
    dim3 grid(N / 128, M / 128);
    gemm_bf16_wmma<<<grid, 256, 0, stream>>>(A, bfB, bias, rowscale, rs_stride,
                                             C, Cb, M, N, K, scale, flags);
  };

  embed_kernel<<<(T_ * H_ + 255) / 256, 256, 0, stream>>>(ids, emb, x);

  for (int l = 0; l < L_; ++l) {
    // ===== MLA =====
    layernorm_kernel<<<T_, 256, 0, stream>>>(x, anw + l * H_, anb + l * H_, xn, xnb);
    cvtW(kvdw + (size_t)l * H_ * KVC_, H_, KVC_);
    gemm(xnb, kvdb + l * KVC_, nullptr, kvb, T_, KVC_, H_, 1.f, 8, nullptr, 0);
    cvtW(kupw + (size_t)l * KVC_ * 1024, KVC_, 1024);
    gemm(kvb, kupb + l * 1024, kc, nullptr, T_, 1024, KVC_, 1.f, 4, nullptr, 0);
    cvtW(vupw + (size_t)l * KVC_ * 1024, KVC_, 1024);
    gemm(kvb, vupb + l * 1024, vv, nullptr, T_, 1024, KVC_, 1.f, 4, nullptr, 0);
    cvtW(krw + (size_t)l * KVC_ * 1024, KVC_, 1024);
    gemm(kvb, krb + l * 1024, kr, nullptr, T_, 1024, KVC_, 1.f, 4, nullptr, 0);
    rope_kernel<<<(T_ * NH_ * 32 + 255) / 256, 256, 0, stream>>>(kr);
    cvtW(qdw + (size_t)l * H_ * QC_, H_, QC_);
    gemm(xnb, qdb + l * QC_, nullptr, qdb_b, T_, QC_, H_, 1.f, 8, nullptr, 0);
    cvtW(qupw + (size_t)l * QC_ * 1024, QC_, 1024);
    gemm(qdb_b, qupb + l * 1024, qc, nullptr, T_, 1024, QC_, 1.f, 4, nullptr, 0);
    cvtW(qrw + (size_t)l * QC_ * 1024, QC_, 1024);
    gemm(qdb_b, qrb_ + l * 1024, qr, nullptr, T_, 1024, QC_, 1.f, 4, nullptr, 0);
    rope_kernel<<<(T_ * NH_ * 32 + 255) / 256, 256, 0, stream>>>(qr);
    mla_attn_kernel<<<B_ * NH_ * S_, 256, 0, stream>>>(qc, qr, kc, kr, vv, ctxb);
    cvtW(outw + (size_t)l * 1024 * H_, 1024, H_);
    gemm(ctxb, outb + l * H_, x, nullptr, T_, H_, 1024, 1.f, 2, nullptr, 0); // x += proj

    // ===== MoE =====
    layernorm_kernel<<<T_, 256, 0, stream>>>(x, mnw + l * H_, mnb + l * H_, xn, xnb);
    moe_gate_kernel<<<T_, 256, 0, stream>>>(xn, gatew + (size_t)l * H_ * E_,
                                            moebias + l * E_, gate);
    fill_zero_kernel<<<(T_ * H_ + 255) / 256, 256, 0, stream>>>(comb, T_ * H_);
    for (int e = 0; e < E_; ++e) {
      cvtW(eupw + ((size_t)l * E_ + e) * H_ * I_, H_, I_);
      gemm(xnb, eupb + ((size_t)l * E_ + e) * I_, nullptr, hmidb,
           T_, I_, H_, 1.f, 8 | 1 /*gelu->bf16*/, nullptr, 0);
      cvtW(ednw + ((size_t)l * E_ + e) * I_ * H_, I_, H_);
      gemm(hmidb, ednb + ((size_t)l * E_ + e) * H_, comb, nullptr,
           T_, H_, I_, 0.1f, 2 /*acc*/, gate + e, E_);
    }
    cvtW(supw + (size_t)l * H_ * I_, H_, I_);
    gemm(xnb, supb + l * I_, nullptr, hmidb, T_, I_, H_, 1.f, 8 | 1, nullptr, 0);
    cvtW(sdnw + (size_t)l * I_ * H_, I_, H_);
    gemm(hmidb, sdnb + l * H_, comb, nullptr, T_, H_, I_, 0.1f, 2, nullptr, 0);
    moe_final_kernel<<<T_, 256, 0, stream>>>(comb, x);
  }

  // ---- final LN + LM head ----
  layernorm_kernel<<<T_, 256, 0, stream>>>(x, fnw, fnb, xn, xnb);
  cvtW(headw, H_, V_);
  gemm(xnb, headb, logits, nullptr, T_, V_, H_, 1.f, 4, nullptr, 0);
}
